// Kmeans_PQ_40243843563651
// MI455X (gfx1250) — matrix operand order repaired and checked by reading, not compiled, not run on hardware
//
#include <hip/hip_runtime.h>
#include <stdint.h>

#define N_VEC   100000
#define M_SUB   8
#define K_CB    256
#define D_SUB   64
#define LEN_VEC (M_SUB * D_SUB)   // 512

typedef _Float16 h16v __attribute__((ext_vector_type(16)));
typedef float    f8v  __attribute__((ext_vector_type(8)));

union Frag {
    h16v     v;
    _Float16 h[16];
    uint32_t u[8];
};

// Block = 256 threads = 8 waves (wave32). Each wave owns one 16-row tile of x.
// LDS: full f16 codebook (256KB) + f32 c_sq (8KB) + per-wave x staging (16KB).
__global__ __launch_bounds__(256) void pq_assign_kernel(
    const float* __restrict__ x,        // [N, 512] f32
    const float* __restrict__ cb,       // [8, 256, 64] f32
    float* __restrict__ qx,             // [N, 512] f32 out
    long long* __restrict__ ids_out)    // [8, N] int64 out
{
    __shared__ __align__(16) _Float16 s_cb[M_SUB * K_CB * D_SUB];  // 256 KB
    __shared__ __align__(16) float    s_csq[M_SUB * K_CB];         //   8 KB
    __shared__ __align__(16) _Float16 s_x[8][16 * D_SUB];          //  16 KB
    __shared__ int                    s_ids[8][16];

    const int tid = threadIdx.x;

    // --- Stage c_sq in exact f32 (deterministic, one thread per codeword row) ---
    #pragma unroll
    for (int r = 0; r < 8; ++r) {
        const int rk = tid * 8 + r;                 // 0..2047
        const float* row = cb + rk * D_SUB;
        float s = 0.f;
        #pragma unroll
        for (int d = 0; d < D_SUB; ++d) s += row[d] * row[d];
        s_csq[rk] = s;
    }
    // --- Stage whole codebook as f16 into LDS (coalesced float4 loads) ---
    for (int it = 0; it < 128; ++it) {
        const int f = it * 256 + tid;               // float4 slot 0..32767
        const float4 v = *(const float4*)(cb + (size_t)f * 4);
        _Float16* d = s_cb + f * 4;
        d[0] = (_Float16)v.x; d[1] = (_Float16)v.y;
        d[2] = (_Float16)v.z; d[3] = (_Float16)v.w;
    }
    __syncthreads();

    const int wave = tid >> 5;
    const int lane = tid & 31;
    const int hi   = lane >> 4;        // half-wave select
    const int ln   = lane & 15;
    const long long tile = (long long)blockIdx.x * 8 + wave;
    const long long n0   = tile * 16;
    if (n0 >= N_VEC) return;           // N divisible by 16; whole-wave guard

    _Float16* xw = s_x[wave];

    for (int m = 0; m < M_SUB; ++m) {
        // --- Prefetch next subspace's x tile (global_prefetch_b8) ---
        if (m < M_SUB - 1) {
            const float* pf = x + (n0 + ln) * (size_t)LEN_VEC
                                + (m + 1) * D_SUB + hi * 32;
            __builtin_prefetch(pf, 0, 1);
        }

        // --- Stage this subspace's 16x64 x tile to LDS as f16 (coalesced) ---
        #pragma unroll
        for (int i = 0; i < 8; ++i) {
            const int f   = i * 32 + lane;          // float4 slot 0..255
            const int row = f >> 4, c4 = f & 15;
            const float4 v = *(const float4*)(x + (n0 + row) * (size_t)LEN_VEC
                                                + m * D_SUB + c4 * 4);
            _Float16* d = xw + row * D_SUB + c4 * 4;
            d[0] = (_Float16)v.x; d[1] = (_Float16)v.y;
            d[2] = (_Float16)v.z; d[3] = (_Float16)v.w;
        }

        // --- Build A fragments (16x32 f16, ISA A-layout K interleave) ---
        Frag a0, a1;
        #pragma unroll
        for (int p = 0; p < 8; ++p) {
            const int kp = ((p < 4) ? 2 * p : 2 * p + 8) + 8 * hi;
            a0.u[p] = *(const uint32_t*)(xw + ln * D_SUB + kp);
            a1.u[p] = *(const uint32_t*)(xw + ln * D_SUB + 32 + kp);
        }

        float bestv[8];
        int   besti[8];
        #pragma unroll
        for (int r = 0; r < 8; ++r) { bestv[r] = 3.4e38f; besti[r] = 0; }

        // --- 16 k-tiles x (2 chained WMMAs over D=64); unroll for pipelining ---
        #pragma unroll 4
        for (int kt = 0; kt < 16; ++kt) {
            const int kc = kt * 16 + ln;            // codeword column
            const _Float16* brow = s_cb + ((size_t)m * K_CB + kc) * D_SUB;
            Frag b0, b1;
            #pragma unroll
            for (int p = 0; p < 8; ++p) {
                const int kp = ((p < 4) ? 2 * p : 2 * p + 8) + 8 * hi;     // B-layout: same K interleave as A
                b0.u[p] = *(const uint32_t*)(brow + kp);
                b1.u[p] = *(const uint32_t*)(brow + 32 + kp);
            }
            f8v acc = {};
            acc = __builtin_amdgcn_wmma_f32_16x16x32_f16(
                false, a0.v, false, b0.v, (short)0, acc, false, false);
            acc = __builtin_amdgcn_wmma_f32_16x16x32_f16(
                false, a1.v, false, b1.v, (short)0, acc, false, false);

            const float cs = s_csq[m * K_CB + kc];
            #pragma unroll
            for (int r = 0; r < 8; ++r) {
                const float dist = cs - 2.0f * acc[r];   // x_sq drops out of argmin
                const bool take = dist < bestv[r];       // kc ascending => first-min
                bestv[r] = take ? dist : bestv[r];
                besti[r] = take ? kc   : besti[r];
            }
        }

        // --- argmin reduce across the 16-lane half (branchless, tie: lowest k) ---
        #pragma unroll
        for (int r = 0; r < 8; ++r) {
            float v = bestv[r]; int bi = besti[r];
            #pragma unroll
            for (int off = 1; off < 16; off <<= 1) {
                const float ov = __shfl_xor(v, off, 32);
                const int   oi = __shfl_xor(bi, off, 32);
                const bool lt = ov < v;
                const bool tie = (ov == v) & (oi < bi);
                const bool take = lt | tie;
                v  = take ? ov : v;
                bi = take ? oi : bi;
            }
            bestv[r] = v; besti[r] = bi;
        }

        // --- lanes 0 / 16 publish ids (rows 0-7 / 8-15) ---
        if (ln == 0) {
            #pragma unroll
            for (int r = 0; r < 8; ++r) {
                const int row = r + 8 * hi;
                s_ids[wave][row] = besti[r];
                ids_out[(size_t)m * N_VEC + n0 + row] = (long long)besti[r];
            }
        }
        // same-wave DS ops are in-order: s_ids visible to all lanes below

        // --- gather exact f32 codeword rows -> Q_x (coalesced float4) ---
        #pragma unroll
        for (int i = 0; i < 8; ++i) {
            const int f   = i * 32 + lane;
            const int row = f >> 4, c4 = f & 15;
            const int id  = s_ids[wave][row];
            const float4 v = *(const float4*)(cb + ((size_t)m * K_CB + id) * D_SUB
                                                 + c4 * 4);
            *(float4*)(qx + (n0 + row) * (size_t)LEN_VEC + m * D_SUB + c4 * 4) = v;
        }
    }
}

extern "C" void kernel_launch(void* const* d_in, const int* in_sizes, int n_in,
                              void* d_out, int out_size, void* d_ws, size_t ws_size,
                              hipStream_t stream) {
    const float* x  = (const float*)d_in[0];
    const float* cb = (const float*)d_in[1];
    float* qx = (float*)d_out;
    long long* ids = (long long*)((char*)d_out
                     + (size_t)N_VEC * LEN_VEC * sizeof(float));
    const int tiles  = (N_VEC + 15) / 16;   // 6250
    const int blocks = (tiles + 7) / 8;     // 782
    pq_assign_kernel<<<blocks, 256, 0, stream>>>(x, cb, qx, ids);
}
